// VishwamAITransformer_31086973288506
// MI455X (gfx1250) — matrix operand order repaired and checked by reading, not compiled, not run on hardware
//
#include <hip/hip_runtime.h>
#include <math.h>

// ---- model dims ----
#define B_    2
#define S_    512
#define D_    1024
#define H_    16
#define E_    4
#define F_    4096
#define KTOP_ 10
#define KD_   256
#define L_    2
#define GATE_ 128
#define DH_   64   // D/H

typedef __attribute__((ext_vector_type(16))) __bf16 v16bf;
typedef __attribute__((ext_vector_type(8)))  float  v8f;

// activation: 0=none 1=gelu(exact) 2=elu+1 3=sigmoid
template<int ACT>
__device__ __forceinline__ float act_fn(float x) {
  if (ACT == 1) return 0.5f * x * (1.0f + erff(x * 0.70710678118654752f));
  if (ACT == 2) return x > 0.0f ? x + 1.0f : expf(x);
  if (ACT == 3) return 1.0f / (1.0f + expf(-x));
  return x;
}

// one 16-byte global -> LDS async copy per lane (ASYNCcnt tracked)
#define ASYNC_CP_B128(ldsoff, gptr)                                        \
  asm volatile("global_load_async_to_lds_b128 %0, %1, off"                 \
               :: "v"(ldsoff), "v"(gptr) : "memory")

// ---------------------------------------------------------------------------
// Generic batched WMMA GEMM:  C[z] = act(alpha * A[z] x B[z] + bias)
//   A: (m,k) at A + z1*sA1 + z2*sA2 + (TA ? k*lda+m : m*lda+k)
//   B: (k,n) at B + z1*sB1 + z2*sB2 + (TB ? n*ldb+k : k*ldb+n)
//   C: (m,n) at C + z1*sC1 + z2*sC2 + m*ldc + n
// Block = 256 threads = 8 waves (4x2 of 16x16 wave tiles) -> 64x32 block tile.
// K stepped by 64: two v_wmma_f32_16x16x32_bf16 per iteration, so barrier +
// asynccnt-wait cost is amortized over 2 matrix ops. f32 tiles are staged
// global->LDS with async-LDS b128 copies (6 per thread per tile, double
// buffered, one s_wait_asynccnt + one barrier per K step) and converted to
// bf16 while building WMMA fragments (v_cvt_pk_bf16_f32); f32 accumulate.
// Requires M%64==0, N%32==0, Kd%64==0 (true for all call sites here).
// LDS tile layouts (f32, 16B-aligned padded rows):
//   TA==0: A[m][k], 64 rows, stride 68   TA==1: A[k][m], 64 rows, stride 68
//   TB==0: B[k][n], 64 rows, stride 36   TB==1: B[n][k], 32 rows, stride 68
// ---------------------------------------------------------------------------
template<int ACT, int TA, int TB>
__global__ __launch_bounds__(256)
void wmma_gemm_kernel(const float* __restrict__ A, const float* __restrict__ Bp,
                      const float* __restrict__ bias, float* __restrict__ C,
                      int M, int N, int Kd, int lda, int ldb, int ldc,
                      int nb1,
                      long long sA1, long long sA2,
                      long long sB1, long long sB2,
                      long long sC1, long long sC2,
                      float alpha)
{
  __shared__ float Asf[2][4352];   // 2 x 17408 B (64 rows x stride 68)
  __shared__ float Bsf[2][2304];   // 2 x  9216 B (64 rows x stride 36 max)

  const int z  = blockIdx.z;
  const int z1 = z % nb1, z2 = z / nb1;
  const float* Az = A  + z1 * sA1 + z2 * sA2;
  const float* Bz = Bp + z1 * sB1 + z2 * sB2;
  float*       Cz = C  + z1 * sC1 + z2 * sC2;

  const int tid  = threadIdx.x;
  const int wave = tid >> 5, lane = tid & 31;
  const int wm = wave & 3, wn = wave >> 2;       // wave tile coords in block
  const int lo = lane & 15, hi = lane >> 4;      // WMMA lane decomposition
  const int m0 = blockIdx.y * 64, n0 = blockIdx.x * 32;

  // ---- loop-invariant staging addresses (4 A chunks + 2 B chunks / thread) --
  // A tile is 64x64 f32 = 1024 16B-chunks; chunk j of this thread:
  //   row r_j = (tid>>4) + 16*j, col c4 = (tid&15)*4, LDS stride 68
  const float *ag0, *ag1, *ag2, *ag3, *bg0, *bg1;
  unsigned aoff0, aoff1, aoff2, aoff3, boff0, boff1;
  {
    const int r = tid >> 4, c4 = (tid & 15) * 4;
    aoff0 = (unsigned)((r +  0) * 68 + c4) * 4u;
    aoff1 = (unsigned)((r + 16) * 68 + c4) * 4u;
    aoff2 = (unsigned)((r + 32) * 68 + c4) * 4u;
    aoff3 = (unsigned)((r + 48) * 68 + c4) * 4u;
    if (TA == 0) {   // row = m, col = k
      ag0 = Az + (long long)(m0 + r +  0) * lda + c4;
      ag1 = Az + (long long)(m0 + r + 16) * lda + c4;
      ag2 = Az + (long long)(m0 + r + 32) * lda + c4;
      ag3 = Az + (long long)(m0 + r + 48) * lda + c4;
    } else {         // row = k, col = m
      ag0 = Az + (long long)(r +  0) * lda + (m0 + c4);
      ag1 = Az + (long long)(r + 16) * lda + (m0 + c4);
      ag2 = Az + (long long)(r + 32) * lda + (m0 + c4);
      ag3 = Az + (long long)(r + 48) * lda + (m0 + c4);
    }
  }
  if (TB == 0) {     // B[k][n]: 64x32 = 512 chunks; k_j = (tid>>3)+32j
    const int k = tid >> 3, nc = (tid & 7) * 4;
    boff0 = (unsigned)((k +  0) * 36 + nc) * 4u;
    boff1 = (unsigned)((k + 32) * 36 + nc) * 4u;
    bg0 = Bz + (long long)(k +  0) * ldb + (n0 + nc);
    bg1 = Bz + (long long)(k + 32) * ldb + (n0 + nc);
  } else {           // B[n][k]: 32x64 = 512 chunks; n_j = (tid>>4)+16j
    const int n = tid >> 4, kc = (tid & 15) * 4;
    boff0 = (unsigned)((n +  0) * 68 + kc) * 4u;
    boff1 = (unsigned)((n + 16) * 68 + kc) * 4u;
    bg0 = Bz + (long long)(n0 + n +  0) * ldb + kc;
    bg1 = Bz + (long long)(n0 + n + 16) * ldb + kc;
  }
  const long long astep = TA ? (long long)64 * lda : 64;   // floats per K step
  const long long bstep = TB ? 64 : (long long)64 * ldb;
  const unsigned abase = (unsigned)(unsigned long long)(void*)&Asf[0][0];
  const unsigned bbase = (unsigned)(unsigned long long)(void*)&Bsf[0][0];

  v8f acc = {};
  const int nk = Kd >> 6;

  // prologue: stage tile 0 into buffer 0
  {
    ASYNC_CP_B128(abase + aoff0, ag0);
    ASYNC_CP_B128(abase + aoff1, ag1);
    ASYNC_CP_B128(abase + aoff2, ag2);
    ASYNC_CP_B128(abase + aoff3, ag3);
    ASYNC_CP_B128(bbase + boff0, bg0);
    ASYNC_CP_B128(bbase + boff1, bg1);
  }

  for (int it = 0; it < nk; ++it) {
    // wait for this wave's copies of tile `it`, then sync the workgroup
    asm volatile("s_wait_asynccnt 0x0" ::: "memory");
    __syncthreads();
    // stage tile it+1 into the other buffer (its last readers finished
    // before the barrier above), overlapping with the compute below
    if (it + 1 < nk) {
      const unsigned ab = abase + (unsigned)(((it + 1) & 1) * 17408);
      const unsigned bb = bbase + (unsigned)(((it + 1) & 1) * 9216);
      const long long s = (long long)(it + 1);
      ASYNC_CP_B128(ab + aoff0, ag0 + s * astep);
      ASYNC_CP_B128(ab + aoff1, ag1 + s * astep);
      ASYNC_CP_B128(ab + aoff2, ag2 + s * astep);
      ASYNC_CP_B128(ab + aoff3, ag3 + s * astep);
      ASYNC_CP_B128(bb + boff0, bg0 + s * bstep);
      ASYNC_CP_B128(bb + boff1, bg1 + s * bstep);
    }
    // ---- two K=32 WMMA halves from the f32 LDS tile (contiguous k = kk + 16*hi + i in A and B) ----
    const float* Ab = &Asf[it & 1][0];
    const float* Bb = &Bsf[it & 1][0];
#pragma unroll
    for (int kk = 0; kk < 64; kk += 32) {
      v16bf afrag, bfrag;
      if (TA == 0) {
        const float* ar = Ab + (wm * 16 + lo) * 68 + kk;
#pragma unroll
        for (int i = 0; i < 8; ++i) afrag[i]     = (__bf16)ar[hi * 16 + i];
#pragma unroll
        for (int i = 0; i < 8; ++i) afrag[8 + i] = (__bf16)ar[hi * 16 + 8 + i];
      } else {
        const int row = wm * 16 + lo;
#pragma unroll
        for (int i = 0; i < 8; ++i) afrag[i]     = (__bf16)Ab[(kk + hi * 16 + i) * 68 + row];
#pragma unroll
        for (int i = 0; i < 8; ++i) afrag[8 + i] = (__bf16)Ab[(kk + hi * 16 + 8 + i) * 68 + row];
      }
      if (TB == 0) {
        const int col = wn * 16 + lo;
#pragma unroll
        for (int i = 0; i < 16; ++i) bfrag[i] = (__bf16)Bb[(kk + hi * 16 + i) * 36 + col];
      } else {
        const float* br = Bb + (wn * 16 + lo) * 68 + kk;
#pragma unroll
        for (int i = 0; i < 16; ++i) bfrag[i] = (__bf16)br[hi * 16 + i];
      }
      acc = __builtin_amdgcn_wmma_f32_16x16x32_bf16(
          /*neg_a=*/false, afrag, /*neg_b=*/false, bfrag,
          /*c_mod=*/(short)0, acc, /*reuse_a=*/false, /*reuse_b=*/false);
    }
  }

  const int gn = n0 + wn * 16 + lo;
  const float bv = bias ? bias[gn] : 0.0f;
#pragma unroll
  for (int i = 0; i < 8; ++i) {
    const int gm = m0 + wm * 16 + hi * 8 + i;
    if (gm < M && gn < N)
      Cz[(long long)gm * ldc + gn] = act_fn<ACT>(alpha * acc[i] + bv);
  }
}

// ---------------------------------------------------------------------------
// Small helper kernels
// ---------------------------------------------------------------------------
template<int ACT>
__global__ void small_gemm_kernel(const float* __restrict__ A, const float* __restrict__ W,
                                  const float* __restrict__ bias, float* __restrict__ C,
                                  int M, int N, int Kd, int lda)
{
  int idx = blockIdx.x * blockDim.x + threadIdx.x;
  if (idx >= M * N) return;
  int m = idx / N, n = idx % N;
  float s = bias ? bias[n] : 0.0f;
  for (int k = 0; k < Kd; ++k) s += A[(long long)m * lda + k] * W[(long long)k * N + n];
  C[idx] = act_fn<ACT>(s);
}

__global__ void softmax_rows_kernel(float* __restrict__ X, int rows, int n)
{
  int wave = threadIdx.x >> 5, lane = threadIdx.x & 31;
  int row = blockIdx.x * (blockDim.x >> 5) + wave;
  if (row >= rows) return;
  float* R = X + (long long)row * n;
  float m = -3.0e38f;
  for (int j = lane; j < n; j += 32) m = fmaxf(m, R[j]);
  for (int o = 16; o; o >>= 1) m = fmaxf(m, __shfl_xor(m, o, 32));
  float s = 0.0f;
  for (int j = lane; j < n; j += 32) s += expf(R[j] - m);
  for (int o = 16; o; o >>= 1) s += __shfl_xor(s, o, 32);
  float inv = 1.0f / s;
  for (int j = lane; j < n; j += 32) R[j] = expf(R[j] - m) * inv;
}

// top-K threshold + masked softmax over rows of 512 (one wave / row)
__global__ void topk_softmax_kernel(float* __restrict__ X, int rows, int ktop)
{
  int wave = threadIdx.x >> 5, lane = threadIdx.x & 31;
  int row = blockIdx.x * (blockDim.x >> 5) + wave;
  if (row >= rows) return;
  float* R = X + (long long)row * 512;
  float orig[16], work[16], pr[16];
#pragma unroll
  for (int j = 0; j < 16; ++j) { orig[j] = R[lane + j * 32]; work[j] = orig[j]; }
  float thr = 0.0f, rowmax = 0.0f;
  for (int t = 0; t < ktop; ++t) {
    float lm = -3.0e38f; int li = 0;
#pragma unroll
    for (int j = 0; j < 16; ++j) if (work[j] > lm) { lm = work[j]; li = j; }
    float wm = lm;
    for (int o = 16; o; o >>= 1) wm = fmaxf(wm, __shfl_xor(wm, o, 32));
    thr = wm;
    if (t == 0) rowmax = wm;
    unsigned long long msk = __ballot(lm == wm);
    int leader = __ffsll((unsigned long long)msk) - 1;
    if (lane == leader) {
#pragma unroll
      for (int j = 0; j < 16; ++j) if (j == li) work[j] = -3.0e38f; // remove one instance
    }
  }
  float s = 0.0f;
#pragma unroll
  for (int j = 0; j < 16; ++j) {
    float v = (orig[j] >= thr) ? orig[j] : -1.0e9f;
    pr[j] = expf(v - rowmax);
    s += pr[j];
  }
  for (int o = 16; o; o >>= 1) s += __shfl_xor(s, o, 32);
  float inv = 1.0f / s;
#pragma unroll
  for (int j = 0; j < 16; ++j) R[lane + j * 32] = pr[j] * inv;
}

// LayerNorm over D=1024, one wave per row, register-cached (in-place safe)
__global__ void layernorm_kernel(const float* __restrict__ In, float* __restrict__ Out,
                                 const float* __restrict__ g, const float* __restrict__ b,
                                 int rows)
{
  int wave = threadIdx.x >> 5, lane = threadIdx.x & 31;
  int row = blockIdx.x * (blockDim.x >> 5) + wave;
  if (row >= rows) return;
  const float* R = In + (long long)row * D_;
  float v[32];
  float s = 0.0f;
#pragma unroll
  for (int j = 0; j < 32; ++j) { v[j] = R[lane + j * 32]; s += v[j]; }
  for (int o = 16; o; o >>= 1) s += __shfl_xor(s, o, 32);
  float mu = s * (1.0f / D_);
  float q = 0.0f;
#pragma unroll
  for (int j = 0; j < 32; ++j) { float d = v[j] - mu; q += d * d; }
  for (int o = 16; o; o >>= 1) q += __shfl_xor(q, o, 32);
  float inv = rsqrtf(q * (1.0f / D_) + 1e-5f);
  float* W = Out + (long long)row * D_;
#pragma unroll
  for (int j = 0; j < 32; ++j) {
    int c = lane + j * 32;
    W[c] = (v[j] - mu) * inv * g[c] + b[c];
  }
}

__global__ void embed_kernel(const int* __restrict__ tok, const float* __restrict__ emb,
                             float* __restrict__ X)
{
  long long idx = (long long)blockIdx.x * blockDim.x + threadIdx.x;
  if (idx >= (long long)B_ * S_ * D_) return;
  int d = (int)(idx % D_);
  long long bs = idx / D_;
  int s = (int)(bs % S_);
  int t = tok[bs];
  float c = -logf(10000.0f) / (float)D_;
  float freq = expf(c * (float)(d & ~1));
  float ang = (float)s * freq;
  float pe = (d & 1) ? cosf(ang) : sinf(ang);
  X[idx] = emb[(long long)t * D_ + d] * 32.0f + pe;  // sqrt(1024)=32
}

__global__ void mean_s_kernel(const float* __restrict__ X, float* __restrict__ avg)
{
  int idx = blockIdx.x * blockDim.x + threadIdx.x;  // B*D
  if (idx >= B_ * D_) return;
  int b = idx / D_, d = idx % D_;
  const float* p = X + (long long)b * S_ * D_ + d;
  float s = 0.0f;
  for (int t = 0; t < S_; ++t) s += p[(long long)t * D_];
  avg[idx] = s * (1.0f / S_);
}

__global__ void moe_accum_kernel(float* __restrict__ moe, const float* __restrict__ eo,
                                 const float* __restrict__ gates, int e, int init)
{
  long long idx = (long long)blockIdx.x * blockDim.x + threadIdx.x;
  if (idx >= (long long)B_ * S_ * D_) return;
  long long bs = idx / D_;
  float g = gates[bs * E_ + e];
  float prev = init ? 0.0f : moe[idx];
  moe[idx] = prev + g * eo[idx];
}

__global__ void zsum_kernel(const float* __restrict__ phik, float* __restrict__ z)
{
  int idx = blockIdx.x * blockDim.x + threadIdx.x;  // B*H*KD
  if (idx >= B_ * H_ * KD_) return;
  int kd = idx % KD_;
  int h = (idx / KD_) % H_;
  int b = idx / (KD_ * H_);
  const float* p = phik + (long long)b * S_ * H_ * KD_ + (long long)h * KD_ + kd;
  float s = 0.0f;
  for (int t = 0; t < S_; ++t) s += p[(long long)t * H_ * KD_];
  z[idx] = s;
}

__global__ void den_kernel(const float* __restrict__ phiq, const float* __restrict__ z,
                           float* __restrict__ den)
{
  int idx = blockIdx.x * blockDim.x + threadIdx.x;  // B*S*H
  if (idx >= B_ * S_ * H_) return;
  int h = idx % H_;
  long long bs = idx / H_;
  int b = (int)(bs / S_);
  const float* pq = phiq + bs * (long long)(H_ * KD_) + (long long)h * KD_;
  const float* pz = z + ((long long)b * H_ + h) * KD_;
  float s = 0.0f;
  for (int k = 0; k < KD_; ++k) s += pq[k] * pz[k];
  den[idx] = s;
}

__global__ void pf_div_kernel(float* __restrict__ num, const float* __restrict__ den)
{
  long long idx = (long long)blockIdx.x * blockDim.x + threadIdx.x;
  if (idx >= (long long)B_ * S_ * D_) return;
  int d = (int)(idx % D_);
  long long bs = idx / D_;
  int h = d / DH_;
  num[idx] = num[idx] / (den[bs * H_ + h] + 1e-6f);
}

__global__ void attn_blend_kernel(const float* __restrict__ x, const float* __restrict__ moe,
                                  const float* __restrict__ sp, const float* __restrict__ pf,
                                  const float* __restrict__ tg, const float* __restrict__ ag,
                                  float depth, float* __restrict__ out)
{
  long long idx = (long long)blockIdx.x * blockDim.x + threadIdx.x;
  if (idx >= (long long)B_ * S_ * D_) return;
  int b = (int)(idx / ((long long)S_ * D_));
  float taa = tg[b * 2 + 0] * sp[idx] + tg[b * 2 + 1] * pf[idx];
  out[idx] = x[idx] + (ag[b * 2 + 0] * moe[idx] + ag[b * 2 + 1] * taa) * depth;
}

__global__ void scale_add_kernel(const float* __restrict__ x, const float* __restrict__ f,
                                 const float* __restrict__ arg, float depth,
                                 float* __restrict__ out)
{
  long long idx = (long long)blockIdx.x * blockDim.x + threadIdx.x;
  if (idx >= (long long)B_ * S_ * D_) return;
  int b = (int)(idx / ((long long)S_ * D_));
  out[idx] = x[idx] + f[idx] * arg[b * 2 + 1] * depth;
}

// ---------------------------------------------------------------------------
// Host side
// ---------------------------------------------------------------------------
static void gemm(hipStream_t st, const float* A, const float* Bp, const float* bias,
                 float* C, int M, int N, int Kd, int lda, int ldb, int ldc,
                 int nb1, int nb2,
                 long long sA1, long long sA2, long long sB1, long long sB2,
                 long long sC1, long long sC2,
                 int transA, int transB, float alpha, int act)
{
  dim3 g(N / 32, M / 64, nb1 * nb2), blk(256);
  if (transA == 1)
    wmma_gemm_kernel<0,1,0><<<g, blk, 0, st>>>(A, Bp, bias, C, M, N, Kd, lda, ldb, ldc,
                                               nb1, sA1, sA2, sB1, sB2, sC1, sC2, alpha);
  else if (transB == 1)
    wmma_gemm_kernel<0,0,1><<<g, blk, 0, st>>>(A, Bp, bias, C, M, N, Kd, lda, ldb, ldc,
                                               nb1, sA1, sA2, sB1, sB2, sC1, sC2, alpha);
  else if (act == 1)
    wmma_gemm_kernel<1,0,0><<<g, blk, 0, st>>>(A, Bp, bias, C, M, N, Kd, lda, ldb, ldc,
                                               nb1, sA1, sA2, sB1, sB2, sC1, sC2, alpha);
  else if (act == 2)
    wmma_gemm_kernel<2,0,0><<<g, blk, 0, st>>>(A, Bp, bias, C, M, N, Kd, lda, ldb, ldc,
                                               nb1, sA1, sA2, sB1, sB2, sC1, sC2, alpha);
  else
    wmma_gemm_kernel<0,0,0><<<g, blk, 0, st>>>(A, Bp, bias, C, M, N, Kd, lda, ldb, ldc,
                                               nb1, sA1, sA2, sB1, sB2, sC1, sC2, alpha);
}

// sorted-key order of each layer's param dict (JAX pytree flattening)
enum {
  I_arg_b1, I_arg_b2, I_arg_w1, I_arg_w2, I_attn_gate_b, I_attn_gate_w,
  I_ffn_b1, I_ffn_b2, I_ffn_w1, I_ffn_w2,
  I_ln1_b, I_ln1_g, I_ln2_b, I_ln2_g, I_ln3_b, I_ln3_g,
  I_moe_bk, I_moe_bo, I_moe_bq, I_moe_bv, I_moe_wk, I_moe_wo, I_moe_wq, I_moe_wv,
  I_pf_bk, I_pf_bo, I_pf_bq, I_pf_bv, I_pf_feat, I_pf_wk, I_pf_wo, I_pf_wq, I_pf_wv,
  I_router_b, I_router_w,
  I_sp_bk, I_sp_bo, I_sp_bq, I_sp_bv, I_sp_wk, I_sp_wo, I_sp_wq, I_sp_wv,
  I_taa_gate_b, I_taa_gate_w, NPAR
};

extern "C" void kernel_launch(void* const* d_in, const int* in_sizes, int n_in,
                              void* d_out, int out_size, void* d_ws, size_t ws_size,
                              hipStream_t stream)
{
  (void)in_sizes; (void)out_size; (void)ws_size;
  const float* embed = (const float*)d_in[0];                 // "params" sorts before "tokens"
  const int*   tokens = (const int*)d_in[n_in - 1];
  const float* lp[L_][NPAR];
  for (int l = 0; l < L_; ++l)
    for (int k = 0; k < NPAR; ++k)
      lp[l][k] = (const float*)d_in[1 + l * NPAR + k];

  // ---- carve workspace ----
  float* W = (float*)d_ws;
  auto take = [&](long long nelem) { float* p = W; W += nelem; return p; };
  const long long BSD = (long long)B_ * S_ * D_;
  const int BS = B_ * S_;
  float* X    = take(BSD);
  float* X2   = take(BSD);
  float* Qb   = take(BSD);
  float* Kb   = take(BSD);
  float* Vb   = take(BSD);
  float* AO   = take(BSD);
  float* PO   = take(BSD);
  float* MOE  = take(BSD);
  float* SP   = take(BSD);
  float* PF   = take(BSD);
  float* SC   = take((long long)B_ * H_ * S_ * S_);
  float* PHQ  = take((long long)BS * H_ * KD_);
  float* PHK  = take((long long)BS * H_ * KD_);
  float* KV   = take((long long)B_ * H_ * KD_ * DH_);
  float* Zb   = take(B_ * H_ * KD_);
  float* DEN  = take(B_ * S_ * H_);
  float* GATES = take(BS * E_);
  float* AVG  = take(B_ * D_);
  float* ARGH = take(B_ * GATE_);
  float* ARG  = take(B_ * 2);
  float* TG   = take(B_ * 2);
  float* AG   = take(B_ * 2);
  float* FFNH = take((long long)BS * F_);

  const long long SD = (long long)S_ * D_;
  const long long SS = (long long)S_ * S_;

  embed_kernel<<<(unsigned)((BSD + 255) / 256), 256, 0, stream>>>(tokens, embed, X);

  for (int li = 0; li < L_; ++li) {
    const float* const* P = lp[li];
    float depth = 1.0f + ((float)li / (float)L_) * 0.5f;

    mean_s_kernel<<<(B_ * D_ + 255) / 256, 256, 0, stream>>>(X, AVG);
    // AdaptiveReasoningGate
    small_gemm_kernel<1><<<(B_ * GATE_ + 255) / 256, 256, 0, stream>>>(
        AVG, P[I_arg_w1], P[I_arg_b1], ARGH, B_, GATE_, D_, D_);
    small_gemm_kernel<3><<<1, 256, 0, stream>>>(
        ARGH, P[I_arg_w2], P[I_arg_b2], ARG, B_, 2, GATE_, GATE_);
    // router gates
    small_gemm_kernel<0><<<(BS * E_ + 255) / 256, 256, 0, stream>>>(
        X, P[I_router_w], P[I_router_b], GATES, BS, E_, D_, D_);
    softmax_rows_kernel<<<(BS + 7) / 8, 256, 0, stream>>>(GATES, BS, E_);

    // ---- MoE expert attention (looped to bound workspace) ----
    for (int e = 0; e < E_; ++e) {
      const long long eW = (long long)e * D_ * D_;
      gemm(stream, X, P[I_moe_wq] + eW, P[I_moe_bq] + e * D_, Qb, BS, D_, D_, D_, D_, D_,
           1, 1, 0, 0, 0, 0, 0, 0, 0, 0, 1.0f, 0);
      gemm(stream, X, P[I_moe_wk] + eW, P[I_moe_bk] + e * D_, Kb, BS, D_, D_, D_, D_, D_,
           1, 1, 0, 0, 0, 0, 0, 0, 0, 0, 1.0f, 0);
      gemm(stream, X, P[I_moe_wv] + eW, P[I_moe_bv] + e * D_, Vb, BS, D_, D_, D_, D_, D_,
           1, 1, 0, 0, 0, 0, 0, 0, 0, 0, 1.0f, 0);
      // scores[b,h] = Q K^T / sqrt(Dh)
      gemm(stream, Qb, Kb, nullptr, SC, S_, S_, DH_, D_, D_, S_,
           H_, B_, DH_, SD, DH_, SD, SS, (long long)H_ * SS, 0, 1, 0.125f, 0);
      softmax_rows_kernel<<<(B_ * H_ * S_ + 7) / 8, 256, 0, stream>>>(SC, B_ * H_ * S_, S_);
      // context = A V
      gemm(stream, SC, Vb, nullptr, AO, S_, DH_, S_, S_, D_, D_,
           H_, B_, SS, (long long)H_ * SS, DH_, SD, DH_, SD, 0, 0, 1.0f, 0);
      gemm(stream, AO, P[I_moe_wo] + eW, P[I_moe_bo] + e * D_, PO, BS, D_, D_, D_, D_, D_,
           1, 1, 0, 0, 0, 0, 0, 0, 0, 0, 1.0f, 0);
      moe_accum_kernel<<<(unsigned)((BSD + 255) / 256), 256, 0, stream>>>(MOE, PO, GATES, e, e == 0);
    }

    // ---- Dynamic sparse attention ----
    gemm(stream, X, P[I_sp_wq], P[I_sp_bq], Qb, BS, D_, D_, D_, D_, D_, 1, 1, 0,0,0,0,0,0, 0,0, 1.0f, 0);
    gemm(stream, X, P[I_sp_wk], P[I_sp_bk], Kb, BS, D_, D_, D_, D_, D_, 1, 1, 0,0,0,0,0,0, 0,0, 1.0f, 0);
    gemm(stream, X, P[I_sp_wv], P[I_sp_bv], Vb, BS, D_, D_, D_, D_, D_, 1, 1, 0,0,0,0,0,0, 0,0, 1.0f, 0);
    gemm(stream, Qb, Kb, nullptr, SC, S_, S_, DH_, D_, D_, S_,
         H_, B_, DH_, SD, DH_, SD, SS, (long long)H_ * SS, 0, 1, 0.125f, 0);
    topk_softmax_kernel<<<(B_ * H_ * S_ + 7) / 8, 256, 0, stream>>>(SC, B_ * H_ * S_, KTOP_);
    gemm(stream, SC, Vb, nullptr, AO, S_, DH_, S_, S_, D_, D_,
         H_, B_, SS, (long long)H_ * SS, DH_, SD, DH_, SD, 0, 0, 1.0f, 0);
    gemm(stream, AO, P[I_sp_wo], P[I_sp_bo], SP, BS, D_, D_, D_, D_, D_, 1, 1, 0,0,0,0,0,0, 0,0, 1.0f, 0);

    // ---- Learned performer attention ----
    gemm(stream, X, P[I_pf_wq], P[I_pf_bq], Qb, BS, D_, D_, D_, D_, D_, 1, 1, 0,0,0,0,0,0, 0,0, 1.0f, 0);
    gemm(stream, X, P[I_pf_wk], P[I_pf_bk], Kb, BS, D_, D_, D_, D_, D_, 1, 1, 0,0,0,0,0,0, 0,0, 1.0f, 0);
    gemm(stream, X, P[I_pf_wv], P[I_pf_bv], Vb, BS, D_, D_, D_, D_, D_, 1, 1, 0,0,0,0,0,0, 0,0, 1.0f, 0);
    // phi = elu(q @ feat)+1  per head
    gemm(stream, Qb, P[I_pf_feat], nullptr, PHQ, BS, KD_, DH_, D_, KD_, H_ * KD_,
         H_, 1, DH_, 0, (long long)DH_ * KD_, 0, KD_, 0, 0, 0, 1.0f, 2);
    gemm(stream, Kb, P[I_pf_feat], nullptr, PHK, BS, KD_, DH_, D_, KD_, H_ * KD_,
         H_, 1, DH_, 0, (long long)DH_ * KD_, 0, KD_, 0, 0, 0, 1.0f, 2);
    zsum_kernel<<<(B_ * H_ * KD_ + 255) / 256, 256, 0, stream>>>(PHK, Zb);
    // kv[b,h] = phiK^T V   (transA)
    gemm(stream, PHK, Vb, nullptr, KV, KD_, DH_, S_, H_ * KD_, D_, DH_,
         H_, B_, KD_, (long long)S_ * H_ * KD_, DH_, SD,
         (long long)KD_ * DH_, (long long)H_ * KD_ * DH_, 1, 0, 1.0f, 0);
    // num[b,h] = phiQ kv
    gemm(stream, PHQ, KV, nullptr, AO, S_, DH_, KD_, H_ * KD_, DH_, D_,
         H_, B_, KD_, (long long)S_ * H_ * KD_,
         (long long)KD_ * DH_, (long long)H_ * KD_ * DH_, DH_, SD, 0, 0, 1.0f, 0);
    den_kernel<<<(B_ * S_ * H_ + 255) / 256, 256, 0, stream>>>(PHQ, Zb, DEN);
    pf_div_kernel<<<(unsigned)((BSD + 255) / 256), 256, 0, stream>>>(AO, DEN);
    gemm(stream, AO, P[I_pf_wo], P[I_pf_bo], PF, BS, D_, D_, D_, D_, D_, 1, 1, 0,0,0,0,0,0, 0,0, 1.0f, 0);

    // ---- gates + blend + residual + LN ----
    small_gemm_kernel<0><<<1, 256, 0, stream>>>(AVG, P[I_taa_gate_w], P[I_taa_gate_b], TG, B_, 2, D_, D_);
    softmax_rows_kernel<<<1, 256, 0, stream>>>(TG, B_, 2);
    small_gemm_kernel<0><<<1, 256, 0, stream>>>(AVG, P[I_attn_gate_w], P[I_attn_gate_b], AG, B_, 2, D_, D_);
    softmax_rows_kernel<<<1, 256, 0, stream>>>(AG, B_, 2);
    attn_blend_kernel<<<(unsigned)((BSD + 255) / 256), 256, 0, stream>>>(
        X, MOE, SP, PF, TG, AG, depth, X2);
    layernorm_kernel<<<(BS + 7) / 8, 256, 0, stream>>>(X2, X, P[I_ln1_g], P[I_ln1_b], BS);

    // ---- FFN ----
    gemm(stream, X, P[I_ffn_w1], P[I_ffn_b1], FFNH, BS, F_, D_, D_, F_, F_,
         1, 1, 0,0,0,0,0,0, 0, 0, 1.0f, 1);  // gelu fused
    gemm(stream, FFNH, P[I_ffn_w2], P[I_ffn_b2], PO, BS, D_, F_, F_, D_, D_,
         1, 1, 0,0,0,0,0,0, 0, 0, 1.0f, 0);
    scale_add_kernel<<<(unsigned)((BSD + 255) / 256), 256, 0, stream>>>(X, PO, ARG, depth, X2);
    layernorm_kernel<<<(BS + 7) / 8, 256, 0, stream>>>(X2, X, P[I_ln2_g], P[I_ln2_b], BS);
    float* dst = (li == L_ - 1) ? (float*)d_out : X;
    layernorm_kernel<<<(BS + 7) / 8, 256, 0, stream>>>(X, dst, P[I_ln3_g], P[I_ln3_b], BS);
  }
}
